// CausalSelfAttention_37641093382444
// MI455X (gfx1250) — compile-verified, matrix operand order checked by reading, not run on hardware
//
#include <hip/hip_runtime.h>
#include <hip/hip_bf16.h>
#include <stdint.h>

// ---------------- problem constants ----------------
#define B_SZ   2
#define T_SZ   2048
#define C_SZ   1024
#define NH     16
#define HD     64
#define BAND   256
#define M_ROWS (B_SZ * T_SZ)      // 4096
#define QTILES (T_SZ / 16)        // 128
#define SW      288               // padded score width (18*16, 9 k-chunks of 32)
#define BLK_M  128
#define BLK_N  128

typedef __attribute__((ext_vector_type(16))) __bf16 v16bf;
typedef __attribute__((ext_vector_type(8)))  __bf16 v8bf;
typedef __attribute__((ext_vector_type(8)))  float  v8f;

// fp32 -> bf16 round-to-nearest-even, no reliance on __bf16 arithmetic
__device__ __forceinline__ __bf16 f2bf(float f) {
    union { float f; uint32_t u; } v; v.f = f;
    uint32_t r = v.u + 0x7FFFu + ((v.u >> 16) & 1u);
    union { unsigned short s; __bf16 b; } o; o.s = (unsigned short)(r >> 16);
    return o.b;
}

__device__ __forceinline__ v8f wmma_bf16(v16bf a, v16bf b, v8f c) {
    return __builtin_amdgcn_wmma_f32_16x16x32_bf16(
        /*neg_a=*/false, a, /*neg_b=*/false, b,
        /*c_mod=*/(short)0, c, /*reuse_a=*/false, /*reuse_b=*/false);
}

// CDNA5 async global->LDS copy (16B per lane), tracked with ASYNCcnt.
__device__ __forceinline__ void async_copy_b128(uint32_t lds_off, const void* gaddr) {
    asm volatile("global_load_async_to_lds_b128 %0, %1, off"
                 :: "v"(lds_off), "v"((uint64_t)(uintptr_t)gaddr)
                 : "memory");
}
__device__ __forceinline__ void wait_async0() {
    asm volatile("s_wait_asynccnt 0x0" ::: "memory");
}

// Load one 16x32 bf16 fragment (row-major source, leading dim `ld` elements).
// CDNA5 16-bit A layout: lane L (m = L&15, half = L>>4) holds
//   elements 0..7  = row m, K = 8*half + 0..7
//   elements 8..15 = row m, K = 16 + 8*half + 0..7
// (B operand of an NT gemm uses the same pattern with lane -> n.)
__device__ __forceinline__ v16bf load_frag16x32(const __bf16* __restrict__ base, int ld) {
    int lane = threadIdx.x & 31;
    int m    = lane & 15;
    int half = lane >> 4;
    const v8bf* p0 = (const v8bf*)(base + (size_t)m * ld + 8 * half);
    const v8bf* p1 = (const v8bf*)(base + (size_t)m * ld + 16 + 8 * half);
    v8bf a0 = *p0;
    v8bf a1 = *p1;
    return __builtin_shufflevector(a0, a1, 0,1,2,3,4,5,6,7,8,9,10,11,12,13,14,15);
}

// ---------------- kernel 0: fp32 -> bf16 convert ----------------
__global__ void cvt_f32_to_bf16(const float* __restrict__ src,
                                __bf16* __restrict__ dst, int n) {
    int i = blockIdx.x * blockDim.x + threadIdx.x;
    if (i < n) dst[i] = f2bf(src[i]);
}

// ---------------- kernel 1: NT GEMM with WMMA + async-LDS B staging ----------
// D[m,n] = sum_k A[m,k] * Bm[n,k] + bias[n]
// Block tile 128x128, 8 waves, each wave owns a 16x128 strip (8 accumulators).
// B tile (128 x 32 bf16, 8 KB) double-buffered in LDS via async global->LDS.
// mode 0: bf16 out, head layout   ((b*NH+h)*T + t)*HD + d   (q, k)
// mode 2: bf16 out, transposed V  ((b*NH+h)*HD + d)*T + t
// mode 3: fp32 out, row-major [m,n]  (final projection)
__global__ __launch_bounds__(256)
void gemm_nt_bf16(const __bf16* __restrict__ A, const __bf16* __restrict__ Bm,
                  const float* __restrict__ bias,
                  __bf16* __restrict__ outb, float* __restrict__ outf,
                  int M, int N, int K, int mode) {
    __shared__ __align__(16) __bf16 sB[2][BLK_N][32];   // 16 KB total

    int tid  = threadIdx.x;
    int wave = tid >> 5;
    int lane = tid & 31;
    int row0 = blockIdx.y * BLK_M + wave * 16;
    int col0 = blockIdx.x * BLK_N;

    v8f acc[8];
    for (int j = 0; j < 8; ++j) acc[j] = (v8f){};

    // stage one 128x32 bf16 B tile into LDS buffer `buf` (async, 32B/thread)
    auto stage = [&](int buf, int kb) {
        for (int s = 0; s < 2; ++s) {
            int o  = tid + s * 256;       // 16-byte unit index, 512 units total
            int n  = o >> 2;              // row within tile (0..127)
            int kq = o & 3;               // 16-byte quarter within row
            uint32_t lds_off = (uint32_t)(uintptr_t)&sB[buf][n][kq * 8];
            const __bf16* g = Bm + (size_t)(col0 + n) * K + kb + kq * 8;
            async_copy_b128(lds_off, g);
        }
    };

    stage(0, 0);
    wait_async0();
    __syncthreads();

    const __bf16* arow = A + (size_t)row0 * K;
    const int nk = K / 32;
    for (int ik = 0; ik < nk; ++ik) {
        int kb  = ik * 32;
        int buf = ik & 1;
        if (ik + 1 < nk) stage(buf ^ 1, kb + 32);   // async prefetch next B tile
        // speculative prefetch of next A chunk -> global_prefetch_b8
        __builtin_prefetch(arow + (size_t)(lane & 15) * K + kb + 32, 0, 0);

        v16bf a = load_frag16x32(arow + kb, K);
        for (int j = 0; j < 8; ++j) {
            v16bf bfrag = load_frag16x32(&sB[buf][j * 16][0], 32);
            acc[j] = wmma_bf16(a, bfrag, acc[j]);
        }
        wait_async0();        // own wave's async writes to LDS done
        __syncthreads();      // everyone's staging visible; safe to swap buffers
    }

    int half = lane >> 4;
    int nn   = lane & 15;
    for (int j = 0; j < 8; ++j) {
        for (int r = 0; r < 8; ++r) {
            int mrow = row0 + r + 8 * half;        // global M index
            int ncol = col0 + j * 16 + nn;         // global N index
            float val = acc[j][r] + bias[ncol];
            if (mode == 3) {
                outf[(size_t)mrow * N + ncol] = val;
            } else {
                int bb = mrow >> 11;               // / T_SZ
                int t  = mrow & (T_SZ - 1);
                int hh = ncol >> 6;                // / HD
                int d  = ncol & (HD - 1);
                if (mode == 2) {
                    outb[((size_t)(bb * NH + hh) * HD + d) * T_SZ + t] = f2bf(val);
                } else {
                    outb[((size_t)(bb * NH + hh) * T_SZ + t) * HD + d] = f2bf(val);
                }
            }
        }
    }
}

// ---------------- kernel 2: banded causal attention ----------------
// one wave (32 threads) per (b, h, 16-row query tile)
__global__ __launch_bounds__(32)
void band_attn(const __bf16* __restrict__ qg, const __bf16* __restrict__ kg,
               const __bf16* __restrict__ vtg, __bf16* __restrict__ yb) {
    __shared__ __align__(16) float  sS[16][SW];
    __shared__ __align__(16) __bf16 sP[16][SW];

    int lane = threadIdx.x & 31;
    int half = lane >> 4;
    int nn   = lane & 15;

    int wg = blockIdx.x;
    int qt = wg & (QTILES - 1);
    int h  = (wg >> 7) & (NH - 1);
    int bi = wg >> 11;
    int i0 = qt * 16;

    int jlo_t = qt - (BAND / 16); if (jlo_t < 0) jlo_t = 0;
    int nkt   = qt - jlo_t + 1;                       // <= 17

    const __bf16* qbase = qg + ((size_t)(bi * NH + h) * T_SZ + i0) * HD;
    v16bf aq0 = load_frag16x32(qbase +  0, HD);
    v16bf aq1 = load_frag16x32(qbase + 32, HD);
    const float scale = 0.125f;                        // 1/sqrt(64)

    // ---- S = Q K^T (masked, scaled) into LDS ----
    for (int kt = 0; kt < nkt; ++kt) {
        int jt = jlo_t + kt;
        const __bf16* kbase = kg + ((size_t)(bi * NH + h) * T_SZ + jt * 16) * HD;
        v16bf bk0 = load_frag16x32(kbase +  0, HD);
        v16bf bk1 = load_frag16x32(kbase + 32, HD);
        v8f c = (v8f){};
        c = wmma_bf16(aq0, bk0, c);
        c = wmma_bf16(aq1, bk1, c);
        for (int r = 0; r < 8; ++r) {
            int mrow = r + 8 * half;
            int i = i0 + mrow;
            int j = jt * 16 + nn;
            bool ok = (j <= i) && (j >= i - BAND);
            sS[mrow][kt * 16 + nn] = ok ? c[r] * scale : -1e30f;
        }
    }
    // pad the remaining columns so softmax weights there are exactly zero
    for (int col = nkt * 16 + lane; col < SW; col += 32)
        for (int mrow = 0; mrow < 16; ++mrow) sS[mrow][col] = -1e30f;
    __syncthreads();

    // ---- per-row softmax (wave32 reductions) ----
    for (int mrow = 0; mrow < 16; ++mrow) {
        float vals[SW / 32];
        float mx = -1e30f;
        for (int t = 0; t < SW / 32; ++t) {
            vals[t] = sS[mrow][t * 32 + lane];
            mx = fmaxf(mx, vals[t]);
        }
        for (int off = 16; off >= 1; off >>= 1)
            mx = fmaxf(mx, __shfl_xor(mx, off, 32));
        float sum = 0.f;
        for (int t = 0; t < SW / 32; ++t) {
            float e = __expf(vals[t] - mx);
            vals[t] = e;
            sum += e;
        }
        for (int off = 16; off >= 1; off >>= 1)
            sum += __shfl_xor(sum, off, 32);
        float inv = 1.0f / sum;                       // diag element always valid
        for (int t = 0; t < SW / 32; ++t)
            sP[mrow][t * 32 + lane] = f2bf(vals[t] * inv);
    }
    __syncthreads();

    // ---- Y = P V  (V stored transposed: [b,h,d,t]) ----
    for (int dt = 0; dt < 4; ++dt) {
        v8f acc = (v8f){};
        const __bf16* vb =
            vtg + ((size_t)(bi * NH + h) * HD + dt * 16) * T_SZ + jlo_t * 16;
        for (int kc = 0; kc < SW / 32; ++kc) {
            // A = P fragment straight out of LDS (same per-lane K pattern)
            int m = nn;
            const v8bf* p0 = (const v8bf*)&sP[m][kc * 32 + 8 * half];
            const v8bf* p1 = (const v8bf*)&sP[m][kc * 32 + 16 + 8 * half];
            v8bf a0 = *p0, a1 = *p1;
            v16bf ap = __builtin_shufflevector(a0, a1,
                        0,1,2,3,4,5,6,7,8,9,10,11,12,13,14,15);
            v16bf bvf = load_frag16x32(vb + kc * 32, T_SZ);
            acc = wmma_bf16(ap, bvf, acc);
        }
        for (int r = 0; r < 8; ++r) {
            int mrow = r + 8 * half;
            size_t row = (size_t)bi * T_SZ + i0 + mrow;
            int col = h * HD + dt * 16 + nn;
            yb[row * C_SZ + col] = f2bf(acc[r]);
        }
    }
}

// ---------------- host-side launch ----------------
extern "C" void kernel_launch(void* const* d_in, const int* in_sizes, int n_in,
                              void* d_out, int out_size, void* d_ws, size_t ws_size,
                              hipStream_t stream) {
    const float* x  = (const float*)d_in[0];
    const float* Wq = (const float*)d_in[1];
    const float* bq = (const float*)d_in[2];
    const float* Wk = (const float*)d_in[3];
    const float* bk = (const float*)d_in[4];
    const float* Wv = (const float*)d_in[5];
    const float* bv = (const float*)d_in[6];
    const float* Wp = (const float*)d_in[7];
    const float* bp = (const float*)d_in[8];
    float* out = (float*)d_out;

    // workspace layout (bytes) -- ~49 MB total
    char* ws = (char*)d_ws;
    __bf16* xb  = (__bf16*)(ws + (size_t)0);                 // 8 MB
    __bf16* Wqb = (__bf16*)(ws + ((size_t)8  << 20));        // 2 MB
    __bf16* Wkb = (__bf16*)(ws + ((size_t)10 << 20));        // 2 MB
    __bf16* Wvb = (__bf16*)(ws + ((size_t)12 << 20));        // 2 MB
    __bf16* Wpb = (__bf16*)(ws + ((size_t)14 << 20));        // 2 MB
    __bf16* qg  = (__bf16*)(ws + ((size_t)16 << 20));        // 8 MB  [b,h,t,d]
    __bf16* kg  = (__bf16*)(ws + ((size_t)24 << 20));        // 8 MB  [b,h,t,d]
    __bf16* vtg = (__bf16*)(ws + ((size_t)32 << 20));        // 8 MB  [b,h,d,t] (+1MB slack)
    __bf16* yb  = (__bf16*)(ws + ((size_t)41 << 20));        // 8 MB  [b*t, c]

    const int nx = M_ROWS * C_SZ;      // 4194304
    const int nw = C_SZ * C_SZ;        // 1048576

    cvt_f32_to_bf16<<<(nx + 255) / 256, 256, 0, stream>>>(x,  xb,  nx);
    cvt_f32_to_bf16<<<(nw + 255) / 256, 256, 0, stream>>>(Wq, Wqb, nw);
    cvt_f32_to_bf16<<<(nw + 255) / 256, 256, 0, stream>>>(Wk, Wkb, nw);
    cvt_f32_to_bf16<<<(nw + 255) / 256, 256, 0, stream>>>(Wv, Wvb, nw);
    cvt_f32_to_bf16<<<(nw + 255) / 256, 256, 0, stream>>>(Wp, Wpb, nw);

    dim3 ggrid(C_SZ / BLK_N, M_ROWS / BLK_M);   // (8, 32)
    dim3 gblock(256);
    gemm_nt_bf16<<<ggrid, gblock, 0, stream>>>(xb, Wqb, bq, qg,  nullptr,
                                               M_ROWS, C_SZ, C_SZ, 0);
    gemm_nt_bf16<<<ggrid, gblock, 0, stream>>>(xb, Wkb, bk, kg,  nullptr,
                                               M_ROWS, C_SZ, C_SZ, 0);
    gemm_nt_bf16<<<ggrid, gblock, 0, stream>>>(xb, Wvb, bv, vtg, nullptr,
                                               M_ROWS, C_SZ, C_SZ, 2);

    band_attn<<<B_SZ * NH * QTILES, 32, 0, stream>>>(qg, kg, vtg, yb);

    gemm_nt_bf16<<<ggrid, gblock, 0, stream>>>(yb, Wpb, bp, nullptr, out,
                                               M_ROWS, C_SZ, C_SZ, 3);
}
